// DeepHaloEncoder_35768487641375
// MI455X (gfx1250) — matrix operand order repaired and checked by reading, not compiled, not run on hardware
//
#include <hip/hip_runtime.h>
#include <hip/hip_bf16.h>

// Problem constants from the reference
#define BTOT 128
#define NSET 4096
#define FF   32
#define EF   16
#define HH   7
#define EHH  112   // E*H

// LDS pitches (halfs) chosen for bank-conflict-free fragment reads
#define PW1  36    // sW1t row pitch: banks 18n -> distinct
#define PW2  116   // sW2t row pitch: banks 58n -> conflict-free vs +8hi
#define PPH  136   // ph row pitch:   banks 4n  -> 2-way worst case

typedef __attribute__((ext_vector_type(16))) _Float16 v16h;
typedef __attribute__((ext_vector_type(8)))  float    v8f;

// A-matrix (16x32 f16) element->K mapping: contiguous k = 16*hi + i (matches B fills).
__device__ __forceinline__ int kpatA(int i, int hi) {
  return i + (hi ? 16 : 0);
}

__device__ __forceinline__ v8f wmma16(v16h a, v16h b, v8f c) {
  return __builtin_amdgcn_wmma_f32_16x16x32_f16(false, a, false, b,
                                                (short)0, c, false, false);
}

// ---------------------------------------------------------------------------
// Encoder: X[t,0:32] -> relu(W1) -> relu(W2) -> W3 -> LN -> Z (f32), Xe (f16)
// One wave per 16-token tile; 4 waves per block. Weights staged transposed
// ([n][k], K zero-padded to 32) in LDS so B fragments are contiguous 32B reads.
// ---------------------------------------------------------------------------
__global__ __launch_bounds__(128) void enc_kernel(
    const float* __restrict__ X,
    const float* __restrict__ W1, const float* __restrict__ b1,
    const float* __restrict__ W2, const float* __restrict__ b2,
    const float* __restrict__ W3, const float* __restrict__ b3,
    const float* __restrict__ g,  const float* __restrict__ be,
    float* __restrict__ Z, _Float16* __restrict__ Xe)
{
  __shared__ _Float16 sWt[3][16][PW1];   // ~3.4 KB: [layer][n][k], k padded to 32
  __shared__ float tile[4][16][17];      // per-wave 16x16 f32 tile (padded)
  const int tid  = threadIdx.x;
  const int lane = tid & 31;
  const int wave = tid >> 5;
  const int l15  = lane & 15;
  const int hi   = lane >> 4;
  const long t0  = ((long)blockIdx.x * 4 + wave) * 16;

  // stage transposed, zero-padded weights (1536 items / 128 thr = 12 iters)
  for (int idx = tid; idx < 3 * 16 * 32; idx += 128) {
    int l = idx >> 9, rem = idx & 511, n = rem >> 5, k = rem & 31;
    const float* W = (l == 0) ? W1 : ((l == 1) ? W2 : W3);
    int kmax = (l == 0) ? FF : EF;
    sWt[l][n][k] = (k < kmax) ? (_Float16)W[k * EF + n] : (_Float16)0.f;
  }
  __syncthreads();

  // ---- layer 1: A = X tile (16x32) ----
  v16h a, wb;
  #pragma unroll
  for (int i = 0; i < 16; ++i)
    a[i] = (_Float16)X[(t0 + l15) * FF + kpatA(i, hi)];
  {
    const _Float16* wr = &sWt[0][l15][16 * hi];
    #pragma unroll
    for (int i = 0; i < 16; ++i) wb[i] = wr[i];
  }
  v8f c1 = {};
  c1 = wmma16(a, wb, c1);
  float bias = b1[l15];
  #pragma unroll
  for (int r = 0; r < 8; ++r) {
    float v = c1[r] + bias;
    tile[wave][r + 8 * hi][l15] = v > 0.f ? v : 0.f;
  }
  asm volatile("s_wait_dscnt 0" ::: "memory");

  // ---- layer 2 (K=16 pad->32; hi lanes are zero) ----
  #pragma unroll
  for (int i = 0; i < 16; ++i)
    a[i] = (hi == 0) ? (_Float16)tile[wave][l15][i] : (_Float16)0.f;
  {
    const _Float16* wr = &sWt[1][l15][16 * hi];
    #pragma unroll
    for (int i = 0; i < 16; ++i) wb[i] = wr[i];
  }
  v8f c2 = {};
  c2 = wmma16(a, wb, c2);
  bias = b2[l15];
  #pragma unroll
  for (int r = 0; r < 8; ++r) {
    float v = c2[r] + bias;
    tile[wave][r + 8 * hi][l15] = v > 0.f ? v : 0.f;
  }
  asm volatile("s_wait_dscnt 0" ::: "memory");

  // ---- layer 3 (no relu) ----
  #pragma unroll
  for (int i = 0; i < 16; ++i)
    a[i] = (hi == 0) ? (_Float16)tile[wave][l15][i] : (_Float16)0.f;
  {
    const _Float16* wr = &sWt[2][l15][16 * hi];
    #pragma unroll
    for (int i = 0; i < 16; ++i) wb[i] = wr[i];
  }
  v8f c3 = {};
  c3 = wmma16(a, wb, c3);
  bias = b3[l15];
  #pragma unroll
  for (int r = 0; r < 8; ++r)
    tile[wave][r + 8 * hi][l15] = c3[r] + bias;
  asm volatile("s_wait_dscnt 0" ::: "memory");

  // ---- LayerNorm over E=16 per token, write Z (f32) + Xe (f16) ----
  if (lane < 16) {
    const long t = t0 + lane;
    float m = 0.f;
    #pragma unroll
    for (int e = 0; e < EF; ++e) m += tile[wave][lane][e];
    m *= (1.f / EF);
    float var = 0.f;
    #pragma unroll
    for (int e = 0; e < EF; ++e) {
      float d = tile[wave][lane][e] - m;
      var += d * d;
    }
    var *= (1.f / EF);
    float inv = rsqrtf(var + 1e-6f);
    #pragma unroll
    for (int e = 0; e < EF; ++e) {
      float z = (tile[wave][lane][e] - m) * inv * g[e] + be[e];
      Z[t * EF + e]  = z;
      Xe[t * EF + e] = (_Float16)z;
    }
  }
}

// ---------------------------------------------------------------------------
// Zbar[b,h] = sum_n af * (Z[b,n,:]^2 @ W_agg[:,h] + b_agg[h]) / max(sum af, 1)
// One block per batch; deterministic LDS tree reduction (no float atomics).
// ---------------------------------------------------------------------------
__global__ __launch_bounds__(256) void zbar_kernel(
    const float* __restrict__ Z, const int* __restrict__ avail,
    const float* __restrict__ Wagg, const float* __restrict__ bagg,
    float* __restrict__ Zbar)
{
  __shared__ float Ws[EF * HH];
  __shared__ float Bs[HH];
  __shared__ float red[256 * 8];
  const int tid = threadIdx.x;
  const int b   = blockIdx.x;
  if (tid < EF * HH) Ws[tid] = Wagg[tid];
  if (tid < HH)      Bs[tid] = bagg[tid];
  __syncthreads();

  float acc[HH] = {};
  float cnt = 0.f;
  for (int n = tid; n < NSET; n += 256) {
    float af = (float)avail[(long)b * NSET + n];
    const float* z = &Z[((long)b * NSET + n) * EF];
    float zs[EF];
    #pragma unroll
    for (int e = 0; e < EF; ++e) { float zz = z[e]; zs[e] = zz * zz; }
    #pragma unroll
    for (int h = 0; h < HH; ++h) {
      float s = Bs[h];
      #pragma unroll
      for (int e = 0; e < EF; ++e) s += zs[e] * Ws[e * HH + h];
      acc[h] += af * s;
    }
    cnt += af;
  }
  #pragma unroll
  for (int h = 0; h < HH; ++h) red[tid * 8 + h] = acc[h];
  red[tid * 8 + 7] = cnt;
  __syncthreads();
  for (int s = 128; s > 0; s >>= 1) {
    if (tid < s) {
      #pragma unroll
      for (int j = 0; j < 8; ++j) red[tid * 8 + j] += red[(tid + s) * 8 + j];
    }
    __syncthreads();
  }
  if (tid < HH) {
    float na = red[7];
    if (na < 1.f) na = 1.f;
    Zbar[b * HH + tid] = red[tid] / na;
  }
}

// ---------------------------------------------------------------------------
// One refinement depth:
//   phi = LN(relu(Xe @ Wphi1 + b1) @ Wphi2 + b2)
//   Z  += einsum('the,h->te', phi.reshape(16,H,E), Zbar[b]) / H
// One wave per 16-token tile; 4 waves per 128-thread block.
// Weights staged transposed in LDS: every B fragment is a contiguous,
// unconditional 32-byte LDS read (conflict-free pitches).
// ---------------------------------------------------------------------------
__global__ __launch_bounds__(128) void update_kernel(
    const _Float16* __restrict__ Xe,
    const float* __restrict__ W1, const float* __restrict__ b1,
    const float* __restrict__ W2, const float* __restrict__ b2,
    const float* __restrict__ g,  const float* __restrict__ be,
    const float* __restrict__ Zbar, float* __restrict__ Z)
{
  __shared__ _Float16 sW1t[EHH * PW1];      // 8064 B  [n][k] k padded to 32
  __shared__ _Float16 sW2t[EHH * PW2];      // 25984 B [n][k] k = 0..111
  __shared__ float    sStage[4][EF * EHH];  // 28672 B per-wave staging (aliased)
  const int tid  = threadIdx.x;
  const int lane = tid & 31, wave = tid >> 5;
  const int l15  = lane & 15, hi = lane >> 4;

  // stage Wphi1 transposed, K zero-padded to 32 (3584 items -> 28 iters)
  for (int idx = tid; idx < EHH * 32; idx += 128) {
    int n = idx >> 5, k = idx & 31;
    sW1t[n * PW1 + k] = (k < EF) ? (_Float16)W1[k * EHH + n] : (_Float16)0.f;
  }
  // stage Wphi2 transposed (coalesced global reads; 12544 items -> 98 iters)
  for (int idx = tid; idx < EHH * EHH; idx += 128) {
    int k = idx / EHH, n = idx - k * EHH;
    sW2t[n * PW2 + k] = (_Float16)W2[k * EHH + n];
  }
  __syncthreads();

  const long t0 = ((long)blockIdx.x * 4 + wave) * 16;
  const long bI = t0 / NSET;                    // batch index (16 | 4096)
  _Float16* ph  = (_Float16*)&sStage[wave][0];  // [16][PPH] f16 phi1 staging

  // zero pad columns 112..127 of ph (read by the c4==3 A chunk)
  for (int idx = lane; idx < 16 * 16; idx += 32)
    ph[(idx >> 4) * PPH + 112 + (idx & 15)] = (_Float16)0.f;

  // ---- phi1: A = Xe tile (16x16 pad->32; hi lanes zero) ----
  v16h a;
  #pragma unroll
  for (int i = 0; i < 16; ++i)
    a[i] = (hi == 0) ? Xe[(t0 + l15) * EF + i] : (_Float16)0.f;
  #pragma unroll
  for (int j = 0; j < HH; ++j) {
    const _Float16* wr = &sW1t[(j * 16 + l15) * PW1 + 16 * hi];
    v16h wb;
    #pragma unroll
    for (int i = 0; i < 16; ++i) wb[i] = wr[i];
    v8f c = {};
    c = wmma16(a, wb, c);
    float bj = b1[j * 16 + l15];
    #pragma unroll
    for (int r = 0; r < 8; ++r) {
      float v = c[r] + bj;
      ph[(r + 8 * hi) * PPH + j * 16 + l15] = (_Float16)(v > 0.f ? v : 0.f);
    }
  }
  asm volatile("s_wait_dscnt 0" ::: "memory");

  // ---- phi2: K=112 (chunks of 32; last chunk zero-masked on hi lanes) ----
  v8f acc[HH] = {};
  #pragma unroll
  for (int c4 = 0; c4 < 4; ++c4) {
    v16h a2;
    #pragma unroll
    for (int i = 0; i < 16; ++i)
      a2[i] = ph[l15 * PPH + c4 * 32 + kpatA(i, hi)];
    #pragma unroll
    for (int j = 0; j < HH; ++j) {
      const int kb = (c4 < 3) ? (c4 * 32 + 16 * hi) : 96;  // c4==3: hi reads masked
      const _Float16* wr = &sW2t[(j * 16 + l15) * PW2 + kb];
      v16h wl;
      #pragma unroll
      for (int i = 0; i < 16; ++i) wl[i] = wr[i];
      v16h wb;
      if (c4 < 3) {
        wb = wl;
      } else {
        #pragma unroll
        for (int i = 0; i < 16; ++i) wb[i] = hi ? (_Float16)0.f : wl[i];
      }
      acc[j] = wmma16(a2, wb, acc[j]);
    }
  }
  asm volatile("s_wait_dscnt 0" ::: "memory");  // ph reads before pf writes

  // ---- bias, stage f32 for LN (reuses the per-wave LDS region) ----
  float* pf = &sStage[wave][0];
  #pragma unroll
  for (int j = 0; j < HH; ++j) {
    float bj = b2[j * 16 + l15];
    #pragma unroll
    for (int r = 0; r < 8; ++r)
      pf[(r + 8 * hi) * EHH + j * 16 + l15] = acc[j][r] + bj;
  }
  asm volatile("s_wait_dscnt 0" ::: "memory");

  // ---- LN over 112, contract with Zbar, Z += delta/H ----
  if (lane < 16) {
    const long t = t0 + lane;
    const float* x = &pf[lane * EHH];
    float m = 0.f;
    for (int k = 0; k < EHH; ++k) m += x[k];
    m *= (1.f / EHH);
    float var = 0.f;
    for (int k = 0; k < EHH; ++k) { float d = x[k] - m; var += d * d; }
    var *= (1.f / EHH);
    float inv = rsqrtf(var + 1e-6f);
    float dlt[EF] = {};
    #pragma unroll
    for (int h = 0; h < HH; ++h) {
      float s = Zbar[bI * HH + h];
      #pragma unroll
      for (int e = 0; e < EF; ++e) {
        int k = h * 16 + e;
        float v = (x[k] - m) * inv * g[k] + be[k];
        dlt[e] += v * s;
      }
    }
    #pragma unroll
    for (int e = 0; e < EF; ++e)
      Z[t * EF + e] += dlt[e] * (1.f / (float)HH);
  }
}

// ---------------------------------------------------------------------------
// out[t, 0:2] = Z[t,:] @ W_proj + b_proj
// ---------------------------------------------------------------------------
__global__ __launch_bounds__(256) void proj_kernel(
    const float* __restrict__ Z, const float* __restrict__ Wp,
    const float* __restrict__ bp, float* __restrict__ out, long total)
{
  long t = (long)blockIdx.x * 256 + threadIdx.x;
  if (t >= total) return;
  float o0 = bp[0], o1 = bp[1];
  #pragma unroll
  for (int e = 0; e < EF; ++e) {
    float z = Z[t * EF + e];
    o0 += z * Wp[e * 2 + 0];
    o1 += z * Wp[e * 2 + 1];
  }
  out[t * 2 + 0] = o0;
  out[t * 2 + 1] = o1;
}

extern "C" void kernel_launch(void* const* d_in, const int* in_sizes, int n_in,
                              void* d_out, int out_size, void* d_ws, size_t ws_size,
                              hipStream_t stream) {
  const float* X      = (const float*)d_in[0];
  const int*   avail  = (const int*)  d_in[1];
  const float* W_enc1 = (const float*)d_in[2];
  const float* b_enc1 = (const float*)d_in[3];
  const float* W_enc2 = (const float*)d_in[4];
  const float* b_enc2 = (const float*)d_in[5];
  const float* W_enc3 = (const float*)d_in[6];
  const float* b_enc3 = (const float*)d_in[7];
  const float* g_enc  = (const float*)d_in[8];
  const float* be_enc = (const float*)d_in[9];
  const float* W_agg  = (const float*)d_in[10];
  const float* b_agg  = (const float*)d_in[11];
  const float* W_phi1 = (const float*)d_in[12];
  const float* b_phi1 = (const float*)d_in[13];
  const float* W_phi2 = (const float*)d_in[14];
  const float* b_phi2 = (const float*)d_in[15];
  const float* g_phi  = (const float*)d_in[16];
  const float* be_phi = (const float*)d_in[17];
  const float* W_proj = (const float*)d_in[18];
  const float* b_proj = (const float*)d_in[19];

  const long tokens = (long)BTOT * NSET;  // 524288

  // Workspace: Z f32 (32MB) | Xe f16 (16MB) | Zbar f32 (3.5KB)
  char* ws = (char*)d_ws;
  float*    Zbuf = (float*)ws;
  _Float16* Xe   = (_Float16*)(ws + tokens * EF * 4);
  float*    Zbar = (float*)(ws + tokens * EF * 4 + tokens * EF * 2);
  if (ws_size < (size_t)(tokens * EF * 6 + BTOT * HH * 4)) return;

  enc_kernel<<<dim3(tokens / 64), dim3(128), 0, stream>>>(
      X, W_enc1, b_enc1, W_enc2, b_enc2, W_enc3, b_enc3, g_enc, be_enc,
      Zbuf, Xe);

  for (int d = 0; d < 3; ++d) {
    zbar_kernel<<<dim3(BTOT), dim3(256), 0, stream>>>(
        Zbuf, avail, W_agg + (long)d * EF * HH, b_agg + (long)d * HH, Zbar);
    update_kernel<<<dim3(tokens / 64), dim3(128), 0, stream>>>(
        Xe,
        W_phi1 + (long)d * EF * EHH,  b_phi1 + (long)d * EHH,
        W_phi2 + (long)d * EHH * EHH, b_phi2 + (long)d * EHH,
        g_phi + (long)d * EHH, be_phi + (long)d * EHH, Zbar, Zbuf);
  }

  proj_kernel<<<dim3(tokens / 256), dim3(256), 0, stream>>>(
      Zbuf, W_proj, b_proj, (float*)d_out, tokens);
}
